// C2AttentionBlock_72756745994849
// MI455X (gfx1250) — matrix operand order repaired and checked by reading, not compiled, not run on hardware
//
#include <hip/hip_runtime.h>

// ============================================================================
// C2AttentionBlock for MI455X (gfx1250, wave32, WMMA).
// One workgroup = one (window, direction): 256 threads = 8 waves = 8 heads.
// bf16 WMMA (v_wmma_f32_16x16x32_bf16) for QK^T, PV, fc1, fc2; f32 softmax/
// GELU/LN. ~288KB dynamic LDS (CDNA5 WGP has 320KB).
//
// Roofline: ~73 GFLOP total vs ~140MB HBM traffic (~6us at 23.3 TB/s) ->
// compute-leaning at bf16 matrix rates; fuse per-window, keep activations in
// LDS, stream bf16-transposed weights from L2 (2MB, resident).
// ============================================================================

typedef __bf16 bf16;
typedef __attribute__((ext_vector_type(8)))  __bf16 v8bf;
typedef __attribute__((ext_vector_type(16))) __bf16 v16bf;
typedef __attribute__((ext_vector_type(8)))  float  v8f;

#define NTOK   64
#define CDIM   256
#define HDIM   1024
#define NHEAD  8
#define HD     32
#define BHWC   8388608   // 8*64*64*256

// ---- LDS layout (bytes) ----------------------------------------------------
//  attention phase:                 MLP phase:
//  [0,32K)    q_bf   64x256         [0,32K)    out_bf 64x256
//  [32K,64K)  kv_bf  64x256         [32K,34K)  LN reduction scratch
//  [64K,96K)  kvT_bf 256x64         [64K,96K)  (free)
//  [96K,160K) out_f32 64x256        [96K,160K) out_f32 (pre-LN accumulator)
//  [160K,288K) S f32 8x(64x64)      [160K,288K) H bf16 64x1024
//  P_bf (8x64x64 bf16, 64K) aliases [0,64K) between QK^T and PV.
#define L_QBF    0
#define L_KVBF   32768
#define L_KVT    65536
#define L_OUTF   98304
#define L_S      163840
#define SMEM_BYTES 294912

// ---- helpers ---------------------------------------------------------------
__device__ __forceinline__ unsigned short f2bfbits(float f) {
  union { float f; unsigned u; } v; v.f = f;
  unsigned r = v.u + 0x7FFFu + ((v.u >> 16) & 1u);   // round-to-nearest-even
  return (unsigned short)(r >> 16);
}
__device__ __forceinline__ bf16 f2bf(float f) {
  union { unsigned short s; bf16 b; } o; o.s = f2bfbits(f); return o.b;
}
__device__ __forceinline__ unsigned pack2bf(float a, float b) {
  return (unsigned)f2bfbits(a) | ((unsigned)f2bfbits(b) << 16);
}
__device__ __forceinline__ v8f vzero8() {
  v8f z;
#pragma unroll
  for (int i = 0; i < 8; ++i) z[i] = 0.0f;
  return z;
}
// A operand (16x32 row-major): lane row M = lane&15;
// elements 0..7 = K(8*half + i), elements 8..15 = K(16 + 8*half + i-8).
__device__ __forceinline__ v16bf load_a(const bf16* rowk, int half) {
  union { v16bf v; v8bf h[2]; } u;
  u.h[0] = *(const v8bf*)(rowk + 8 * half);
  u.h[1] = *(const v8bf*)(rowk + 16 + 8 * half);
  return u.v;
}
// B operand (32x16): lane = column N;
// elements 0..7 = K(8*half + i), elements 8..15 = K(16 + 8*half + i-8).
__device__ __forceinline__ v16bf load_b(const bf16* rowk, int half) {
  union { v16bf v; v8bf h[2]; } u;
  u.h[0] = *(const v8bf*)(rowk + 8 * half);
  u.h[1] = *(const v8bf*)(rowk + 16 + 8 * half);
  return u.v;
}
__device__ __forceinline__ v8f wmma_bf(v16bf a, v16bf b, v8f c) {
  return __builtin_amdgcn_wmma_f32_16x16x32_bf16(false, a, false, b,
                                                 (short)0, c, false, false);
}
// window (win, token) -> flat spatial row index into [B*H*W]
__device__ __forceinline__ int wrow(int win, int tok) {
  int b  = win >> 6, hb = (win >> 3) & 7, wb = win & 7;
  int i  = tok >> 3, j  = tok & 7;
  return ((b << 6) + (hb << 3) + i) * 64 + (wb << 3) + j;
}

// ---- weight prep: f32 -> transposed bf16 (W1T [2][1024][256], W2T [2][256][1024])
__global__ void c2attn_prep_weights(const float* __restrict__ f1w0,
                                    const float* __restrict__ f2w0,
                                    const float* __restrict__ f1w1,
                                    const float* __restrict__ f2w1,
                                    bf16* __restrict__ w1t,
                                    bf16* __restrict__ w2t) {
  unsigned idx   = blockIdx.x * 256u + threadIdx.x;   // 1<<20 total
  unsigned which = idx >> 18;
  unsigned e     = idx & 0x3FFFFu;
  if (which == 0)      { unsigned n = e >> 8,  k = e & 255;  w1t[e]          = f2bf(f1w0[k * 1024 + n]); }
  else if (which == 1) { unsigned n = e >> 10, k = e & 1023; w2t[e]          = f2bf(f2w0[k * 256  + n]); }
  else if (which == 2) { unsigned n = e >> 8,  k = e & 255;  w1t[262144 + e] = f2bf(f1w1[k * 1024 + n]); }
  else                 { unsigned n = e >> 10, k = e & 1023; w2t[262144 + e] = f2bf(f2w1[k * 256  + n]); }
}

// ---- main fused kernel -----------------------------------------------------
__global__ __launch_bounds__(256) void c2attn_win_kernel(
    const float* __restrict__ c1, const float* __restrict__ c2,
    const bf16*  __restrict__ w1t_all, const bf16* __restrict__ w2t_all,
    const float* __restrict__ b1_0, const float* __restrict__ b2_0,
    const float* __restrict__ b1_1, const float* __restrict__ b2_1,
    const float* __restrict__ g0, const float* __restrict__ be0,
    const float* __restrict__ g1, const float* __restrict__ be1,
    float* __restrict__ out) {
  extern __shared__ char smem[];
  bf16*  q_bf  = (bf16*)(smem + L_QBF);
  bf16*  kv_bf = (bf16*)(smem + L_KVBF);
  bf16*  kvT   = (bf16*)(smem + L_KVT);
  float* outf  = (float*)(smem + L_OUTF);
  float* Sall  = (float*)(smem + L_S);
  bf16*  Pall  = (bf16*)(smem + L_QBF);    // aliases q_bf/kv_bf after QK^T
  bf16*  outbf = (bf16*)(smem + L_QBF);    // aliases P after PV
  bf16*  Hl    = (bf16*)(smem + L_S);      // aliases S in MLP phase
  float* red   = (float*)(smem + L_KVBF);  // LN scratch in MLP phase

  const int tid  = threadIdx.x;
  const int wv   = tid >> 5;         // wave == head
  const int lane = tid & 31;
  const int half = lane >> 4;
  const int ln16 = lane & 15;

  const int blk = blockIdx.x;        // 0..1023
  const int dir = blk >> 9;
  const int win = blk & 511;

  const float* qs  = dir ? c2 : c1;
  const float* ks  = dir ? c1 : c2;
  const bf16*  w1t = w1t_all + (size_t)dir * 262144;
  const bf16*  w2t = w2t_all + (size_t)dir * 262144;
  const float* b1  = dir ? b1_1 : b1_0;
  const float* b2  = dir ? b2_1 : b2_0;
  const float* lng = dir ? g1 : g0;
  const float* lnb = dir ? be1 : be0;
  float*       dst = out + (size_t)dir * BHWC;

  // ---- phase 1: load q / kv, convert to bf16; kv also transposed ----------
#pragma unroll
  for (int it = 0; it < 16; ++it) {
    int lin = tid + (it << 8);                 // 0..4095 (64 tok x 64 float4)
    int tok = lin >> 6;
    int c4  = (lin & 63) << 2;
    size_t roff = (size_t)wrow(win, tok) * CDIM + c4;
    float4 qv = *(const float4*)(qs + roff);
    float4 kq = *(const float4*)(ks + roff);
    int base = tok * CDIM + c4;
    *(unsigned*)(q_bf + base)      = pack2bf(qv.x, qv.y);
    *(unsigned*)(q_bf + base + 2)  = pack2bf(qv.z, qv.w);
    *(unsigned*)(kv_bf + base)     = pack2bf(kq.x, kq.y);
    *(unsigned*)(kv_bf + base + 2) = pack2bf(kq.z, kq.w);
    kvT[(c4 + 0) * NTOK + tok] = f2bf(kq.x);
    kvT[(c4 + 1) * NTOK + tok] = f2bf(kq.y);
    kvT[(c4 + 2) * NTOK + tok] = f2bf(kq.z);
    kvT[(c4 + 3) * NTOK + tok] = f2bf(kq.w);
  }
  __syncthreads();

  // ---- phase 2: S = Q K^T / sqrt(hd), one head per wave -------------------
  // Batch ALL operand loads, accumulate 16 tiles in registers, then one
  // store burst: avoids LDS load/store alias serialization around WMMA.
  const int hbase = wv * HD;
  float* Sh = Sall + wv * 4096;
  {
    v16bf aQ[4], bK[4];
#pragma unroll
    for (int tm = 0; tm < 4; ++tm)
      aQ[tm] = load_a(q_bf + (tm * 16 + ln16) * CDIM + hbase, half);
#pragma unroll
    for (int tn = 0; tn < 4; ++tn)
      bK[tn] = load_b(kv_bf + (tn * 16 + ln16) * CDIM + hbase, half);
    v8f sc[4][4];
#pragma unroll
    for (int tn = 0; tn < 4; ++tn)
#pragma unroll
      for (int tm = 0; tm < 4; ++tm)
        sc[tm][tn] = wmma_bf(aQ[tm], bK[tn], vzero8());
#pragma unroll
    for (int tm = 0; tm < 4; ++tm)
#pragma unroll
      for (int tn = 0; tn < 4; ++tn)
#pragma unroll
        for (int r = 0; r < 8; ++r)
          Sh[(tm * 16 + r + 8 * half) * 64 + tn * 16 + ln16] =
              sc[tm][tn][r] * 0.17677669529663689f;   // 1/sqrt(32)
  }
  __syncthreads();

  // ---- phase 3: row softmax (f32 in regs), emit P as bf16 -----------------
  bf16* Ph = Pall + wv * 4096;
#pragma unroll
  for (int rr = lane; rr < 64; rr += 32) {
    float4 buf[16];
    float mx = -3.4e38f;
#pragma unroll
    for (int i = 0; i < 16; ++i) {
      buf[i] = *(const float4*)(Sh + rr * 64 + i * 4);
      mx = fmaxf(mx, fmaxf(fmaxf(buf[i].x, buf[i].y), fmaxf(buf[i].z, buf[i].w)));
    }
    float s = 0.0f;
#pragma unroll
    for (int i = 0; i < 16; ++i) {
      buf[i].x = __expf(buf[i].x - mx);
      buf[i].y = __expf(buf[i].y - mx);
      buf[i].z = __expf(buf[i].z - mx);
      buf[i].w = __expf(buf[i].w - mx);
      s += buf[i].x + buf[i].y + buf[i].z + buf[i].w;
    }
    float inv = 1.0f / s;
#pragma unroll
    for (int i = 0; i < 16; ++i) {
      uint2 p;
      p.x = pack2bf(buf[i].x * inv, buf[i].y * inv);
      p.y = pack2bf(buf[i].z * inv, buf[i].w * inv);
      *(uint2*)(Ph + rr * 64 + i * 4) = p;
    }
  }
  __syncthreads();

  // ---- phase 4: O = P V  (64x32 per head), batched loads ------------------
  {
    v16bf aP[4][2], bV[2][2];
#pragma unroll
    for (int tm = 0; tm < 4; ++tm)
#pragma unroll
      for (int kc = 0; kc < 2; ++kc)
        aP[tm][kc] = load_a(Ph + (tm * 16 + ln16) * 64 + kc * 32, half);
#pragma unroll
    for (int tn = 0; tn < 2; ++tn)
#pragma unroll
      for (int kc = 0; kc < 2; ++kc)
        bV[tn][kc] = load_b(kvT + (hbase + tn * 16 + ln16) * NTOK + kc * 32, half);
    v8f oc[4][2];
#pragma unroll
    for (int tn = 0; tn < 2; ++tn)
#pragma unroll
      for (int tm = 0; tm < 4; ++tm) {
        v8f t = wmma_bf(aP[tm][0], bV[tn][0], vzero8());
        oc[tm][tn] = wmma_bf(aP[tm][1], bV[tn][1], t);
      }
#pragma unroll
    for (int tn = 0; tn < 2; ++tn)
#pragma unroll
      for (int tm = 0; tm < 4; ++tm)
#pragma unroll
        for (int r = 0; r < 8; ++r)
          outf[(tm * 16 + r + 8 * half) * CDIM + hbase + tn * 16 + ln16] =
              oc[tm][tn][r];
  }
  __syncthreads();

  // ---- phase 5a: out f32 -> bf16 (MLP A-operand), aliases P region --------
#pragma unroll
  for (int it = 0; it < 16; ++it) {
    int idx = (tid + (it << 8)) << 2;          // float4-granular
    float4 v = *(const float4*)(outf + idx);
    uint2 p;
    p.x = pack2bf(v.x, v.y);
    p.y = pack2bf(v.z, v.w);
    *(uint2*)(outbf + idx) = p;
  }
  __syncthreads();

  // ---- phase 5b: fc1 + bias + exact GELU -> H (bf16) ----------------------
  // wave wv owns columns [wv*128, wv*128+128) in four 32-col groups;
  // A (LDS) and B (global, L2-resident) double-buffered across k-chunks.
  for (int g = 0; g < 4; ++g) {
    const int colbase = wv * 128 + g * 32;
    v8f acc[4][2];
#pragma unroll
    for (int tm = 0; tm < 4; ++tm)
#pragma unroll
      for (int tn = 0; tn < 2; ++tn) acc[tm][tn] = vzero8();
    v16bf a[4], b[2];
#pragma unroll
    for (int tm = 0; tm < 4; ++tm)
      a[tm] = load_a(outbf + (tm * 16 + ln16) * CDIM, half);
#pragma unroll
    for (int tn = 0; tn < 2; ++tn)
      b[tn] = load_b(w1t + (colbase + tn * 16 + ln16) * CDIM, half);
    for (int kc = 0; kc < 8; ++kc) {
      const int kn = (kc + 1) & 7;             // wrap: last preload is benign
      v16bf an[4], bn[2];
#pragma unroll
      for (int tm = 0; tm < 4; ++tm)
        an[tm] = load_a(outbf + (tm * 16 + ln16) * CDIM + kn * 32, half);
#pragma unroll
      for (int tn = 0; tn < 2; ++tn) {
        const bf16* wp = w1t + (colbase + tn * 16 + ln16) * CDIM + kn * 32;
        __builtin_prefetch(wp + 32, 0, 1);
        bn[tn] = load_b(wp, half);
      }
#pragma unroll
      for (int tn = 0; tn < 2; ++tn)
#pragma unroll
        for (int tm = 0; tm < 4; ++tm)
          acc[tm][tn] = wmma_bf(a[tm], b[tn], acc[tm][tn]);
#pragma unroll
      for (int tm = 0; tm < 4; ++tm) a[tm] = an[tm];
#pragma unroll
      for (int tn = 0; tn < 2; ++tn) b[tn] = bn[tn];
    }
#pragma unroll
    for (int tn = 0; tn < 2; ++tn) {
      int col = colbase + tn * 16 + ln16;
      float bias = b1[col];
#pragma unroll
      for (int tm = 0; tm < 4; ++tm)
#pragma unroll
        for (int r = 0; r < 8; ++r) {
          float x = acc[tm][tn][r] + bias;
          float ge = 0.5f * x * (1.0f + erff(x * 0.70710678118654752f));
          Hl[(tm * 16 + r + 8 * half) * HDIM + col] = f2bf(ge);
        }
    }
  }
  __syncthreads();

  // ---- phase 6: fc2 + bias + attn-out + input residual -> pre-LN (f32 LDS)
  {
    const int colbase = wv * 32;
    v8f acc[4][2];
#pragma unroll
    for (int tm = 0; tm < 4; ++tm)
#pragma unroll
      for (int tn = 0; tn < 2; ++tn) acc[tm][tn] = vzero8();
    v16bf a[4], b[2];
#pragma unroll
    for (int tm = 0; tm < 4; ++tm)
      a[tm] = load_a(Hl + (tm * 16 + ln16) * HDIM, half);
#pragma unroll
    for (int tn = 0; tn < 2; ++tn)
      b[tn] = load_b(w2t + (colbase + tn * 16 + ln16) * HDIM, half);
    for (int kc = 0; kc < 32; ++kc) {
      const int kn = (kc + 1) & 31;            // wrap: last preload is benign
      v16bf an[4], bn[2];
#pragma unroll
      for (int tm = 0; tm < 4; ++tm)
        an[tm] = load_a(Hl + (tm * 16 + ln16) * HDIM + kn * 32, half);
#pragma unroll
      for (int tn = 0; tn < 2; ++tn) {
        const bf16* wp = w2t + (colbase + tn * 16 + ln16) * HDIM + kn * 32;
        __builtin_prefetch(wp + 32, 0, 1);
        bn[tn] = load_b(wp, half);
      }
#pragma unroll
      for (int tn = 0; tn < 2; ++tn)
#pragma unroll
        for (int tm = 0; tm < 4; ++tm)
          acc[tm][tn] = wmma_bf(a[tm], b[tn], acc[tm][tn]);
#pragma unroll
      for (int tm = 0; tm < 4; ++tm) a[tm] = an[tm];
#pragma unroll
      for (int tn = 0; tn < 2; ++tn) b[tn] = bn[tn];
    }
#pragma unroll
    for (int tn = 0; tn < 2; ++tn) {
      int col = colbase + tn * 16 + ln16;
      float bias = b2[col];
#pragma unroll
      for (int tm = 0; tm < 4; ++tm)
#pragma unroll
        for (int r = 0; r < 8; ++r) {
          int row = tm * 16 + r + 8 * half;
          float v = acc[tm][tn][r] + bias + outf[row * CDIM + col] +
                    qs[(size_t)wrow(win, row) * CDIM + col];
          outf[row * CDIM + col] = v;
        }
    }
  }
  __syncthreads();

  // ---- phase 7: LayerNorm over C=256 per token, un-windowed store ---------
  {
    int tok = tid >> 2, part = tid & 3, cb = part * 64;
    float4 tb[16];
    float s = 0.0f, s2 = 0.0f;
#pragma unroll
    for (int i = 0; i < 16; ++i) {
      tb[i] = *(const float4*)(outf + tok * CDIM + cb + i * 4);
      s  += tb[i].x + tb[i].y + tb[i].z + tb[i].w;
      s2 += tb[i].x * tb[i].x + tb[i].y * tb[i].y +
            tb[i].z * tb[i].z + tb[i].w * tb[i].w;
    }
    red[tok * 8 + part]     = s;
    red[tok * 8 + 4 + part] = s2;
    __syncthreads();
    float ts = red[tok * 8] + red[tok * 8 + 1] + red[tok * 8 + 2] + red[tok * 8 + 3];
    float tq = red[tok * 8 + 4] + red[tok * 8 + 5] + red[tok * 8 + 6] + red[tok * 8 + 7];
    float mu   = ts * (1.0f / 256.0f);
    float var  = tq * (1.0f / 256.0f) - mu * mu;
    float rstd = rsqrtf(var + 1e-5f);
    size_t obase = (size_t)wrow(win, tok) * CDIM + cb;
#pragma unroll
    for (int i = 0; i < 16; ++i) {
      float4 gv = *(const float4*)(lng + cb + i * 4);
      float4 bv = *(const float4*)(lnb + cb + i * 4);
      float4 o;
      o.x = (tb[i].x - mu) * rstd * gv.x + bv.x;
      o.y = (tb[i].y - mu) * rstd * gv.y + bv.y;
      o.z = (tb[i].z - mu) * rstd * gv.z + bv.z;
      o.w = (tb[i].w - mu) * rstd * gv.w + bv.w;
      *(float4*)(dst + obase + i * 4) = o;
    }
  }
}

// ============================================================================
extern "C" void kernel_launch(void* const* d_in, const int* in_sizes, int n_in,
                              void* d_out, int out_size, void* d_ws, size_t ws_size,
                              hipStream_t stream) {
  const float* c1    = (const float*)d_in[0];
  const float* c2    = (const float*)d_in[1];
  // d_in[2] = window_size (==8, compiled in)
  const float* m1f1w = (const float*)d_in[3];
  const float* m1f1b = (const float*)d_in[4];
  const float* m1f2w = (const float*)d_in[5];
  const float* m1f2b = (const float*)d_in[6];
  const float* ln1g  = (const float*)d_in[7];
  const float* ln1b  = (const float*)d_in[8];
  const float* m2f1w = (const float*)d_in[9];
  const float* m2f1b = (const float*)d_in[10];
  const float* m2f2w = (const float*)d_in[11];
  const float* m2f2b = (const float*)d_in[12];
  const float* ln2g  = (const float*)d_in[13];
  const float* ln2b  = (const float*)d_in[14];

  bf16* w1t = (bf16*)d_ws;                                   // 2 x 1024x256 bf16 (1MB)
  bf16* w2t = (bf16*)((char*)d_ws + (size_t)2 * 262144 * 2); // 2 x 256x1024 bf16 (1MB)

  c2attn_prep_weights<<<4096, 256, 0, stream>>>(m1f1w, m1f2w, m2f1w, m2f2w, w1t, w2t);

  (void)hipFuncSetAttribute((const void*)c2attn_win_kernel,
                            hipFuncAttributeMaxDynamicSharedMemorySize, SMEM_BYTES);
  c2attn_win_kernel<<<dim3(1024), dim3(256), SMEM_BYTES, stream>>>(
      c1, c2, w1t, w2t, m1f1b, m1f2b, m2f1b, m2f2b,
      ln1g, ln1b, ln2g, ln2b, (float*)d_out);
}
